// VocoderNetwork_76381698392918
// MI455X (gfx1250) — compile-verified, matrix operand order checked by reading, not run on hardware
//
#include <hip/hip_runtime.h>

#define NBLK 4
#define NLAY 9
#define NWIN 800
#define FCH  128
#define MGCD 60
#define RFLD 512
#define SQF  0.70710678118f

typedef __attribute__((ext_vector_type(16))) __bf16       v16bf;
typedef __attribute__((ext_vector_type(8)))  float        v8f;
typedef __attribute__((ext_vector_type(4)))  unsigned int u32x4;
typedef __attribute__((ext_vector_type(4)))  float        f32x4;

struct Q2 { u32x4 a, b; };   // 32 bytes == v16bf

__device__ __forceinline__ unsigned short f2bf(float f) {
  unsigned u = __builtin_bit_cast(unsigned, f);
  u += 0x7fffu + ((u >> 16) & 1u);            // round-to-nearest-even
  return (unsigned short)(u >> 16);
}
__device__ __forceinline__ float bf2f(unsigned short h) {
  return __builtin_bit_cast(float, (unsigned)h << 16);
}

// ---------------------------------------------------------------- init
__global__ void k_init(const float* __restrict__ sig, float* __restrict__ prevbuf,
                       float* __restrict__ out) {
  int i = blockIdx.x * blockDim.x + threadIdx.x;
  if (i < NWIN + RFLD) {
    prevbuf[i] = sig[i];
    if (i >= RFLD) out[i - RFLD] = sig[i];    // signal[RF:] output
  }
}

// ------------------------------------------------- cond = tanh(mgc @ W.T + b)
__global__ void k_cond(const float* __restrict__ mgc, const float* __restrict__ cw,
                       const float* __restrict__ cb, float* __restrict__ cond) {
  int i = blockIdx.x * blockDim.x + threadIdx.x;
  if (i >= NWIN * MGCD) return;
  int n = i / MGCD, m = i - n * MGCD;
  int t = n / 200;
  int f = (n % 200) * MGCD + m;
  const float* wr = cw + (size_t)f * MGCD;
  const float* mr = mgc + t * MGCD;
  float acc = cb[f];
  for (int d = 0; d < MGCD; ++d) acc += mr[d] * wr[d];
  cond[i] = tanhf(acc);
}

// ---------------- per-(blk,layer) bias tables: cond-linear + conv biases folded
__global__ void k_condbias(const float* __restrict__ cond, const float* __restrict__ clw,
                           const float* __restrict__ clb, const float* __restrict__ c0b,
                           const float* __restrict__ cLb, float* __restrict__ cib,
                           float* __restrict__ cgb) {
  int i = blockIdx.x * blockDim.x + threadIdx.x;  // NBLK*NLAY*NWIN*FCH exact
  int o  = i & (FCH - 1);
  int n  = (i >> 7) % NWIN;
  int bl = i / (FCH * NWIN);                      // blk*NLAY + l
  int blk = bl / NLAY, l = bl - blk * NLAY;
  const float* cn = cond + n * MGCD;
  const float* w0 = clw + ((size_t)(bl * 2 + 0) * FCH + o) * MGCD;
  const float* w1 = clw + ((size_t)(bl * 2 + 1) * FCH + o) * MGCD;
  float a0 = clb[(bl * 2 + 0) * FCH + o];
  float a1 = clb[(bl * 2 + 1) * FCH + o];
  for (int d = 0; d < MGCD; ++d) { float c = cn[d]; a0 += c * w0[d]; a1 += c * w1[d]; }
  if (l == 0) {
    a0 += c0b[(blk * 3 + 0) * FCH + o];
    a1 += c0b[(blk * 3 + 1) * FCH + o];
  } else {
    int bm = (blk * 8 + l - 1) * 3;
    a0 += cLb[(bm + 0) * FCH + o];
    a1 += cLb[(bm + 1) * FCH + o];
  }
  cib[i] = a0; cgb[i] = a1;
}

// -------- repack convL_w [NB,8,3,F,F,2] fp32 -> [NB*8][t*128+o][k*128+c] bf16
__global__ void k_wpack(const float* __restrict__ wL, unsigned short* __restrict__ wp) {
  int i  = blockIdx.x * blockDim.x + threadIdx.x;  // NBLK*8*384*256 exact
  int kk = i & 255;
  int m  = (i >> 8) % 384;
  int bl = i / (384 * 256);
  int t = m >> 7, o = m & 127;
  int k = kk >> 7, c = kk & 127;
  size_t src = ((((size_t)bl * 3 + t) * FCH + o) * FCH + c) * 2 + k;
  wp[i] = f2bf(wL[src]);
}

// ------------------------------------------------- layer 0 (Cin=1), VALU
__global__ void k_layer0(const float* __restrict__ prevbuf, const float* __restrict__ w0,
                         const float* __restrict__ cib, const float* __restrict__ cgb,
                         const float* __restrict__ b2, unsigned short* __restrict__ hout) {
  int idx = blockIdx.x * blockDim.x + threadIdx.x;  // NWIN*256*8 exact
  int og = idx & 7;
  int t2 = idx >> 3;
  int j = t2 & 255;
  int n = t2 >> 8;
  float s0 = prevbuf[n + 2 * j], s1 = prevbuf[n + 2 * j + 1];
  int ob = og * 16;
  const float* ci = cib + (size_t)n * FCH + ob;
  const float* cg = cgb + (size_t)n * FCH + ob;
  unsigned short res[16];
#pragma unroll
  for (int r = 0; r < 16; ++r) {
    int o = ob + r;
    float d0 = w0[(0 * FCH + o) * 2 + 0] * s0 + w0[(0 * FCH + o) * 2 + 1] * s1 + ci[r];
    float d1 = w0[(1 * FCH + o) * 2 + 0] * s0 + w0[(1 * FCH + o) * 2 + 1] * s1 + cg[r];
    float d2 = w0[(2 * FCH + o) * 2 + 0] * s0 + w0[(2 * FCH + o) * 2 + 1] * s1 + b2[o];
    float it = tanhf(d0);
    float gt = 1.f / (1.f + expf(-d1));
    res[r] = f2bf((it * gt + d2) * SQF);
  }
  unsigned pk[8];
#pragma unroll
  for (int q = 0; q < 8; ++q) pk[q] = (unsigned)res[2 * q] | ((unsigned)res[2 * q + 1] << 16);
  u32x4* dst = (u32x4*)(hout + ((size_t)(n * 256 + j)) * FCH + ob);
  u32x4 v0 = {pk[0], pk[1], pk[2], pk[3]}, v1 = {pk[4], pk[5], pk[6], pk[7]};
  dst[0] = v0; dst[1] = v1;
}

// ------------------- gated CondConv layer as WMMA GEMM (layers 1..8)
// One wave32 per block. Each wave computes COLS column tiles x one 16-channel
// tile, with all three gate rows (input/gate/residual) so gating stays
// in-register. A fragments are reused across COLS column tiles.
template <int COLS>
__global__ __launch_bounds__(32) void k_condconv(
    const unsigned short* __restrict__ wp,    // [384][256] bf16 (t*128+o, k*128+c)
    const unsigned short* __restrict__ hin,   // [N][2*Lhalf][128] bf16
    unsigned short* __restrict__ hout,        // [N][Lhalf][128] bf16
    const float* __restrict__ cib, const float* __restrict__ cgb,
    const float* __restrict__ b2, int Lhalf) {
  int lane = threadIdx.x;
  int ot = blockIdx.y;                         // channel tile 0..7
  int kh = (lane & 16) ? 8 : 0;                // per-lane K/M half per ISA layout

  int nn[COLS], jj[COLS];
  const unsigned short* bcol[COLS];
#pragma unroll
  for (int c = 0; c < COLS; ++c) {
    int col = (blockIdx.x * COLS + c) * 16 + (lane & 15);   // WMMA N index
    int n = col / Lhalf;
    int j = col - n * Lhalf;
    nn[c] = n; jj[c] = j;
    bcol[c] = hin + ((size_t)n * 2 * Lhalf + 2 * j) * FCH + kh;
  }
  const unsigned short* arow = wp + ((size_t)(ot * 16 + (lane & 15))) * 256 + kh;

  v8f acc[3][COLS];
#pragma unroll
  for (int t = 0; t < 3; ++t)
#pragma unroll
    for (int c = 0; c < COLS; ++c) acc[t][c] = (v8f){};

#pragma unroll
  for (int s = 0; s < 8; ++s) {                // K = 256 -> 8 x k32
    v16bf fb[COLS];
#pragma unroll
    for (int c = 0; c < COLS; ++c) {
      const u32x4* bp = (const u32x4*)(bcol[c] + s * 32);
      Q2 q; q.a = bp[0]; q.b = bp[2];
      fb[c] = __builtin_bit_cast(v16bf, q);
    }
#pragma unroll
    for (int t = 0; t < 3; ++t) {
      const u32x4* ap = (const u32x4*)(arow + (size_t)t * 128 * 256 + s * 32);
      Q2 qa; qa.a = ap[0]; qa.b = ap[2];
      v16bf fa = __builtin_bit_cast(v16bf, qa);
#pragma unroll
      for (int c = 0; c < COLS; ++c)
        acc[t][c] = __builtin_amdgcn_wmma_f32_16x16x32_bf16(
            false, fa, false, fb[c], (short)0, acc[t][c], false, false);
    }
  }

  int ob = ot * 16 + kh;                       // lane's 8 consecutive channels
  float b2v[8];
  *(f32x4*)&b2v[0] = *(const f32x4*)(b2 + ob);
  *(f32x4*)&b2v[4] = *(const f32x4*)(b2 + ob + 4);

#pragma unroll
  for (int c = 0; c < COLS; ++c) {
    int n = nn[c], j = jj[c];
    float civ[8], cgv[8];
    *(f32x4*)&civ[0] = *(const f32x4*)(cib + (size_t)n * FCH + ob);
    *(f32x4*)&civ[4] = *(const f32x4*)(cib + (size_t)n * FCH + ob + 4);
    *(f32x4*)&cgv[0] = *(const f32x4*)(cgb + (size_t)n * FCH + ob);
    *(f32x4*)&cgv[4] = *(const f32x4*)(cgb + (size_t)n * FCH + ob + 4);

    unsigned short res[8];
#pragma unroll
    for (int r = 0; r < 8; ++r) {
      float d0 = acc[0][c][r] + civ[r];
      float d1 = acc[1][c][r] + cgv[r];
      float d2 = acc[2][c][r] + b2v[r];
      float it = tanhf(d0);
      float gt = 1.f / (1.f + expf(-d1));
      res[r] = f2bf((it * gt + d2) * SQF);
    }
    unsigned pk[4];
#pragma unroll
    for (int q = 0; q < 4; ++q)
      pk[q] = (unsigned)res[2 * q] | ((unsigned)res[2 * q + 1] << 16);
    u32x4 v = {pk[0], pk[1], pk[2], pk[3]};
    *(u32x4*)(hout + ((size_t)n * Lhalf + j) * FCH + ob) = v;
  }
}

// ------------------- head: pre=relu(h@pre_w.T+b); mean/logvar; reparameterize
__global__ __launch_bounds__(256) void k_head(
    const unsigned short* __restrict__ hfin, const float* __restrict__ pw,
    const float* __restrict__ pb, const float* __restrict__ mw, const float* __restrict__ mb,
    const float* __restrict__ sw, const float* __restrict__ sb, const float* __restrict__ eps,
    float* __restrict__ prevbuf, float* __restrict__ out, int blk) {
  __shared__ float hs[FCH];
  __shared__ float r1[256], r2[256];
  int n = blockIdx.x, tid = threadIdx.x;
  if (tid < FCH) hs[tid] = bf2f(hfin[(size_t)n * FCH + tid]);
  __syncthreads();
  const float* wrow = pw + (size_t)tid * FCH;
  float acc = pb[tid];
  for (int c = 0; c < FCH; ++c) acc += hs[c] * wrow[c];
  float pre = fmaxf(acc, 0.f);
  r1[tid] = pre * mw[tid];
  r2[tid] = pre * sw[tid];
  __syncthreads();
  for (int st = 128; st > 0; st >>= 1) {
    if (tid < st) { r1[tid] += r1[tid + st]; r2[tid] += r2[tid + st]; }
    __syncthreads();
  }
  if (tid == 0) {
    float mean = r1[0] + mb[0];
    float logv = r2[0] + sb[0];
    float nx = eps[n] * expf(0.5f * logv) + mean;
    prevbuf[RFLD + n] = nx;                              // next block's input
    if (blk == NBLK - 1) { out[NWIN + n] = mean; out[2 * NWIN + n] = logv; }
    if (n >= NWIN - RFLD) out[3 * NWIN + blk * RFLD + (n - (NWIN - RFLD))] = nx;
  }
}

extern "C" void kernel_launch(void* const* d_in, const int* in_sizes, int n_in,
                              void* d_out, int out_size, void* d_ws, size_t ws_size,
                              hipStream_t stream) {
  const float* mgc       = (const float*)d_in[0];
  const float* signal    = (const float*)d_in[1];
  const float* cond_w    = (const float*)d_in[2];
  const float* cond_b    = (const float*)d_in[3];
  const float* conv0_w   = (const float*)d_in[4];
  const float* conv0_b   = (const float*)d_in[5];
  const float* convL_w   = (const float*)d_in[6];
  const float* convL_b   = (const float*)d_in[7];
  const float* condlin_w = (const float*)d_in[8];
  const float* condlin_b = (const float*)d_in[9];
  const float* pre_w     = (const float*)d_in[10];
  const float* pre_b     = (const float*)d_in[11];
  const float* mean_w    = (const float*)d_in[12];
  const float* mean_b    = (const float*)d_in[13];
  const float* std_w     = (const float*)d_in[14];
  const float* std_b     = (const float*)d_in[15];
  const float* eps       = (const float*)d_in[16];
  float* out = (float*)d_out;

  char* ws = (char*)d_ws;
  size_t off = 0;
  auto take = [&](size_t bytes) -> char* {
    char* p = ws + off;
    off += (bytes + 255) & ~(size_t)255;
    return p;
  };
  float*          cond    = (float*)take((size_t)NWIN * MGCD * 4);
  float*          cib     = (float*)take((size_t)NBLK * NLAY * NWIN * FCH * 4);
  float*          cgb     = (float*)take((size_t)NBLK * NLAY * NWIN * FCH * 4);
  unsigned short* wpack   = (unsigned short*)take((size_t)NBLK * 8 * 384 * 256 * 2);
  unsigned short* hA      = (unsigned short*)take((size_t)NWIN * 256 * FCH * 2);
  unsigned short* hB      = (unsigned short*)take((size_t)NWIN * 128 * FCH * 2);
  float*          prevbuf = (float*)take((size_t)(NWIN + RFLD) * 4);
  if (off > ws_size) return;

  k_init<<<(NWIN + RFLD + 255) / 256, 256, 0, stream>>>(signal, prevbuf, out);
  k_cond<<<(NWIN * MGCD + 255) / 256, 256, 0, stream>>>(mgc, cond_w, cond_b, cond);
  k_condbias<<<NBLK * NLAY * NWIN * FCH / 256, 256, 0, stream>>>(
      cond, condlin_w, condlin_b, conv0_b, convL_b, cib, cgb);
  k_wpack<<<NBLK * 8 * 384 * 256 / 256, 256, 0, stream>>>(convL_w, wpack);

  for (int blk = 0; blk < NBLK; ++blk) {
    const float* cib_blk = cib + (size_t)(blk * NLAY) * NWIN * FCH;
    const float* cgb_blk = cgb + (size_t)(blk * NLAY) * NWIN * FCH;

    k_layer0<<<NWIN * 256 * 8 / 256, 256, 0, stream>>>(
        prevbuf, conv0_w + blk * 3 * FCH * 2,
        cib_blk, cgb_blk, conv0_b + (blk * 3 + 2) * FCH, hA);

    unsigned short* hin = hA;
    unsigned short* hout = hB;
    for (int l = 1; l <= 8; ++l) {
      int Lh = 256 >> l;
      int tiles = NWIN * Lh / 16;
      const unsigned short* wpl = wpack + (size_t)(blk * 8 + l - 1) * 384 * 256;
      const float* cibl = cib_blk + (size_t)l * NWIN * FCH;
      const float* cgbl = cgb_blk + (size_t)l * NWIN * FCH;
      const float* b2l  = convL_b + (size_t)((blk * 8 + l - 1) * 3 + 2) * FCH;
      if (tiles % 4 == 0) {
        dim3 grid(tiles / 4, 8);
        k_condconv<4><<<grid, 32, 0, stream>>>(wpl, hin, hout, cibl, cgbl, b2l, Lh);
      } else {
        dim3 grid(tiles / 2, 8);
        k_condconv<2><<<grid, 32, 0, stream>>>(wpl, hin, hout, cibl, cgbl, b2l, Lh);
      }
      unsigned short* t = hin; hin = hout; hout = t;
    }

    k_head<<<NWIN, 256, 0, stream>>>(
        hin, pre_w + (size_t)blk * 256 * FCH, pre_b + blk * 256,
        mean_w + blk * 256, mean_b + blk, std_w + blk * 256, std_b + blk,
        eps + blk * NWIN, prevbuf, out, blk);
  }
}
